// BasicLSTM_37297495998596
// MI455X (gfx1250) — matrix operand order repaired and checked by reading, not compiled, not run on hardware
//
#include <hip/hip_runtime.h>

// ---------------------------------------------------------------------------
// BasicLSTM for MI455X (gfx1250).
//   bf16 WMMA GEMMs (weights pre-converted to bf16 once per launch) with
//   double-buffered LDS staging of weight tiles via async-to-LDS DMA
//   (ASYNCcnt), + fused LSTM-step kernels.  Operand loads for a whole stage
//   are grouped ahead of a sched_barrier so 8 WMMAs issue back-to-back
//   behind a single wait.
//   H=768, L=10, T(W)=10, B=512.  All matmuls X @ W^T, W (out,in) row-major.
// ---------------------------------------------------------------------------

#define HDIM   768
#define LAYERS 10
#define TSTEPS 10
#define BATCH  512
#define G4H    (4 * HDIM)

#define KSTAGE  64   // k-elements staged per LDS buffer
#define BROWS   64   // weight rows (N columns of C) staged per block
#define LDS_ROW 72   // padded row stride in bf16 elems (144B): 16 lanes ->
                     // dword strides l*36 mod 64 all distinct (conflict-free)

typedef __attribute__((ext_vector_type(16))) __bf16 v16bf;
typedef __attribute__((ext_vector_type(8)))  __bf16 v8bf;
typedef __attribute__((ext_vector_type(8)))  float  v8f;
typedef __attribute__((ext_vector_type(4)))  float  v4f;
typedef __attribute__((vector_size(16)))     int    v4i;   // native int4 vector

// Async global->LDS path (ASYNCcnt-tracked DMA) if the toolchain declares it.
// Probe-confirmed signature: (v4i __device__*, v4i __shared__*, int, int).
#if defined(__gfx1250__) && __has_builtin(__builtin_amdgcn_global_load_async_to_lds_b128) && __has_builtin(__builtin_amdgcn_s_wait_asynccnt)
#define USE_ASYNC_LDS 1
typedef __attribute__((address_space(1))) v4i gv4i;
typedef __attribute__((address_space(3))) v4i lv4i;
#else
#define USE_ASYNC_LDS 0
#endif

#if defined(__gfx1250__) && __has_builtin(__builtin_amdgcn_sched_barrier)
#define SCHED_FENCE() __builtin_amdgcn_sched_barrier(0)
#else
#define SCHED_FENCE()
#endif

struct StageRegs { v4i r0, r1; };

// Issue one thread's share of a stage: 2x 16B from global row -> LDS row.
// Async: fire both DMA copies now (overlap with compute). Sync: load to regs.
__device__ __forceinline__ void stage_issue(const __bf16* g, __bf16* l,
                                            StageRegs& s) {
#if USE_ASYNC_LDS
  v4i* g0 = (v4i*)(void*)const_cast<__bf16*>(g);
  v4i* g1 = (v4i*)(void*)const_cast<__bf16*>(g + 32);
  v4i* l0 = (v4i*)(void*)l;
  v4i* l1 = (v4i*)(void*)(l + 32);
  __builtin_amdgcn_global_load_async_to_lds_b128((gv4i*)g0, (lv4i*)l0, 0, 0);
  __builtin_amdgcn_global_load_async_to_lds_b128((gv4i*)g1, (lv4i*)l1, 0, 0);
  (void)s;
#else
  s.r0 = *(const v4i*)g;
  s.r1 = *(const v4i*)(g + 32);
#endif
}

// Complete the stage before the barrier. Async: drain ASYNCcnt. Sync: ds_store.
__device__ __forceinline__ void stage_commit(__bf16* l, StageRegs& s) {
#if USE_ASYNC_LDS
  __builtin_amdgcn_s_wait_asynccnt(0);
  (void)l; (void)s;
#else
  *(v4i*)l        = s.r0;
  *(v4i*)(l + 32) = s.r1;
#endif
}

// ---- A fragment (16x32 bf16, ISA 7.12.2 layout) ---------------------------
// lane L: m = L&15, half = L>>4.
// elements 0..7  = row[8*half + 0..7]        (VGPR0-3)
// elements 8..15 = row[16 + 8*half + 0..7]   (VGPR4-7)
__device__ __forceinline__ v16bf load_a_frag_bf16(const __bf16* __restrict__ row,
                                                  int half) {
  v8bf lo = *(const v8bf*)(row + 8 * half);
  v8bf hi = *(const v8bf*)(row + 16 + 8 * half);
  v16bf r;
#pragma unroll
  for (int i = 0; i < 8; ++i) { r[i] = lo[i]; r[i + 8] = hi[i]; }
  return r;
}

// ---- B fragment (32x16 bf16) from an LDS-staged bf16 row ------------------
// lane L: n = L&15, half = L>>4; elements 0..7 = row[8*half + 0..7],
// elements 8..15 = row[16 + 8*half + 0..7].
__device__ __forceinline__ v16bf load_b_frag_lds(const __bf16* row, int half) {
  v8bf lo = *(const v8bf*)(row + 8 * half);
  v8bf hi = *(const v8bf*)(row + 16 + 8 * half);
  v16bf r;
#pragma unroll
  for (int i = 0; i < 8; ++i) { r[i] = lo[i]; r[i + 8] = hi[i]; }
  return r;
}

__device__ __forceinline__ float sigm(float x) {
  return 1.0f / (1.0f + __expf(-x));
}

// ---------------------------------------------------------------------------
// fp32 -> bf16 weight conversion, 8 elems per thread (2x b128 in, 1x b128 out)
// ---------------------------------------------------------------------------
__global__ __launch_bounds__(256) void convert_w_kernel(
    const float* __restrict__ src, __bf16* __restrict__ dst) {
  long i = ((long)blockIdx.x * 256 + threadIdx.x) * 8;
  v4f a = *(const v4f*)(src + i);
  v4f b = *(const v4f*)(src + i + 4);
  v8bf o;
#pragma unroll
  for (int j = 0; j < 4; ++j) {
    o[j]     = (__bf16)a[j];
    o[4 + j] = (__bf16)b[j];
  }
  *(v8bf*)(dst + i) = o;
}

// ---------------------------------------------------------------------------
// (b,t,h) -> (t,b,h) transpose + fp32->bf16 convert of x_position.
// ---------------------------------------------------------------------------
__global__ __launch_bounds__(256) void reorder_x_kernel(
    const float* __restrict__ x, __bf16* __restrict__ out) {
  long idx = (long)blockIdx.x * 256 + threadIdx.x;
  int  h   = (int)(idx % HDIM);
  long r   = idx / HDIM;
  int  b   = (int)(r % BATCH);
  int  t   = (int)(r / BATCH);
  out[idx] = (__bf16)x[((long)b * TSTEPS + t) * HDIM + h];
}

// ---------------------------------------------------------------------------
// GEMM:  C[M,N] = A_bf16[M,K] @ W_bf16[N,K]^T + b1 + b2
// Block = 256 threads (8 waves). Block tile: M=128 (wave w -> rows m0+16w),
// N=64 (4 WMMA tiles per wave). B strip staged via LDS (async DMA), double-
// buffered in KSTAGE=64 chunks; next-stage copies overlap the 8 WMMAs.
// Per stage: load 2 A frags + 8 B frags, sched-fence, then 8 WMMAs.
// Chunked-A addressing: A(m,k) = A[chunkBase(k/chunkK) + m*rowStride + k%chunkK]
//   (row-major: chunkK=K, chunkOuter=0; fc1 gather: chunkK=H, chunkOuter=B*H).
// ---------------------------------------------------------------------------
__global__ __launch_bounds__(256) void gemm_bias_kernel(
    const __bf16* __restrict__ A, const __bf16* __restrict__ W,
    const float* __restrict__ b1, const float* __restrict__ b2,
    void* __restrict__ Cout, int M, int N, int K,
    int rowStride, int chunkK, long chunkOuter, int outBf16) {
  __shared__ __bf16 lds[2 * BROWS * LDS_ROW];  // 18,432 B

  const int tid  = threadIdx.x;
  const int wave = tid >> 5, lane = tid & 31;
  const int half = lane >> 4, l16 = lane & 15;
  const int m0 = blockIdx.x * 128 + wave * 16;
  const int n0 = blockIdx.y * 64;

  // Staging role: thread -> (row, 16B segment); each thread moves 2x 16B.
  const int srow = tid >> 2, sseg = tid & 3;
  const __bf16* wsrc = W + (long)(n0 + srow) * K + sseg * 8;
  const int     ldst = srow * LDS_ROW + sseg * 8;  // elem offset within buffer

  v8f acc[4] = {};
  const long abase     = (long)(m0 + l16) * rowStride;
  long       chunkBase = 0;
  int        kin       = 0;

  {  // Prologue: stage k0 = 0 into buffer 0.
    StageRegs sr;
    stage_issue(wsrc, &lds[ldst], sr);
    stage_commit(&lds[ldst], sr);
  }

  int cur = 0;
  for (int k0 = 0; k0 < K; k0 += KSTAGE) {
    __syncthreads();  // buf[cur] visible; all reads of buf[cur^1] done
    const bool more = (k0 + KSTAGE) < K;
    StageRegs sr;
    __bf16* lnext = &lds[(cur ^ 1) * (BROWS * LDS_ROW) + ldst];
    if (more) stage_issue(wsrc + (k0 + KSTAGE), lnext, sr);

    // ---- load ALL operands for this stage, then fence, then 8 WMMAs ----
    const __bf16* bbase = &lds[cur * (BROWS * LDS_ROW)];
    const __bf16* arowp = A + chunkBase + abase + kin;
    v16bf af0 = load_a_frag_bf16(arowp, half);
    v16bf af1 = load_a_frag_bf16(arowp + 32, half);
    v16bf b0[4], b1[4];
#pragma unroll
    for (int tt = 0; tt < 4; ++tt) {
      const __bf16* brow = bbase + (16 * tt + l16) * LDS_ROW;
      b0[tt] = load_b_frag_lds(brow, half);
      b1[tt] = load_b_frag_lds(brow + 32, half);
    }
    SCHED_FENCE();
#pragma unroll
    for (int tt = 0; tt < 4; ++tt)
      acc[tt] = __builtin_amdgcn_wmma_f32_16x16x32_bf16(
          false, af0, false, b0[tt], (short)0, acc[tt], false, false);
#pragma unroll
    for (int tt = 0; tt < 4; ++tt)
      acc[tt] = __builtin_amdgcn_wmma_f32_16x16x32_bf16(
          false, af1, false, b1[tt], (short)0, acc[tt], false, false);

    if (more) stage_commit(lnext, sr);
    kin += KSTAGE;
    if (kin == chunkK) { kin = 0; chunkBase += chunkOuter; }
    cur ^= 1;
  }

#pragma unroll
  for (int tt = 0; tt < 4; ++tt) {
    const int n = n0 + 16 * tt + l16;
    float bias = (b1 ? b1[n] : 0.0f) + (b2 ? b2[n] : 0.0f);
#pragma unroll
    for (int r = 0; r < 8; ++r) {
      const int m = m0 + r + 8 * half;
      float v = acc[tt][r] + bias;
      if (outBf16) ((__bf16*)Cout)[(long)m * N + n] = (__bf16)v;
      else         ((float*)Cout)[(long)m * N + n]  = v;
    }
  }
}

// ---------------------------------------------------------------------------
// Fused recurrent step:  g = gx[t] + h_{t-1} @ w_hh^T ; gates ; c,h update.
// Each wave owns a 16x16 (m,n) tile and computes the 4 matching gate tiles
// (i,f,g,o at n, H+n, 2H+n, 3H+n) -> nonlinearities fuse locally.
// The staged B strip = rows {g*H + n0 + 0..15 | g=0..3} of w_hh (bf16).
// h_{t-1} read from ys[t-1], h_t written to ys[t] -> no cross-block races.
// Block = 8 waves => 128 batch rows x 16 h-cols. Grid (B/128, H/16).
// ---------------------------------------------------------------------------
__global__ __launch_bounds__(256) void lstm_step_kernel(
    const float*  __restrict__ gx,    // (T*B, 4H) this layer
    const __bf16* __restrict__ whh,   // (4H, H)   this layer, bf16
    __bf16* __restrict__ ybuf,        // (T*B, H)  this layer's h outputs
    float*  __restrict__ cbuf,        // (B, H)
    int t) {
  __shared__ __bf16 lds[2 * BROWS * LDS_ROW];

  const int tid  = threadIdx.x;
  const int wave = tid >> 5, lane = tid & 31;
  const int half = lane >> 4, l16 = lane & 15;
  const int m0 = blockIdx.x * 128 + wave * 16;
  const int n0 = blockIdx.y * 16;

  v8f acc[4] = {};
  if (t > 0) {
    // staging: strip row r = gate*16 + rr -> w_hh row gate*H + n0 + rr
    const int srow = tid >> 2, sseg = tid & 3;
    const int wrow = (srow >> 4) * HDIM + n0 + (srow & 15);
    const __bf16* wsrc = whh + (long)wrow * HDIM + sseg * 8;
    const int     ldst = srow * LDS_ROW + sseg * 8;

    const __bf16* arow =
        ybuf + (long)(t - 1) * BATCH * HDIM + (long)(m0 + l16) * HDIM;

    {
      StageRegs sr;
      stage_issue(wsrc, &lds[ldst], sr);
      stage_commit(&lds[ldst], sr);
    }
    int cur = 0;
    for (int k0 = 0; k0 < HDIM; k0 += KSTAGE) {
      __syncthreads();
      const bool more = (k0 + KSTAGE) < HDIM;
      StageRegs sr;
      __bf16* lnext = &lds[(cur ^ 1) * (BROWS * LDS_ROW) + ldst];
      if (more) stage_issue(wsrc + (k0 + KSTAGE), lnext, sr);

      const __bf16* bbase = &lds[cur * (BROWS * LDS_ROW)];
      v16bf af0 = load_a_frag_bf16(arow + k0, half);
      v16bf af1 = load_a_frag_bf16(arow + k0 + 32, half);
      v16bf b0[4], b1[4];
#pragma unroll
      for (int g = 0; g < 4; ++g) {
        const __bf16* brow = bbase + (16 * g + l16) * LDS_ROW;
        b0[g] = load_b_frag_lds(brow, half);
        b1[g] = load_b_frag_lds(brow + 32, half);
      }
      SCHED_FENCE();
#pragma unroll
      for (int g = 0; g < 4; ++g)
        acc[g] = __builtin_amdgcn_wmma_f32_16x16x32_bf16(
            false, af0, false, b0[g], (short)0, acc[g], false, false);
#pragma unroll
      for (int g = 0; g < 4; ++g)
        acc[g] = __builtin_amdgcn_wmma_f32_16x16x32_bf16(
            false, af1, false, b1[g], (short)0, acc[g], false, false);

      if (more) stage_commit(lnext, sr);
      cur ^= 1;
    }
  }

  const int n = n0 + l16;
#pragma unroll
  for (int r = 0; r < 8; ++r) {
    const int m = m0 + r + 8 * half;
    const float* gr = gx + (long)(t * BATCH + m) * G4H;
    float gi = acc[0][r] + gr[n];
    float gf = acc[1][r] + gr[HDIM + n];
    float gg = acc[2][r] + gr[2 * HDIM + n];
    float go = acc[3][r] + gr[3 * HDIM + n];
    float iv = sigm(gi);
    float fv = sigm(gf);
    float gv = tanhf(gg);
    float ov = sigm(go);
    float cold = (t > 0) ? cbuf[(long)m * HDIM + n] : 0.0f;
    float cn = fv * cold + iv * gv;
    float hv = ov * tanhf(cn);
    cbuf[(long)m * HDIM + n] = cn;
    ybuf[(long)(t * BATCH + m) * HDIM + n] = (__bf16)hv;
  }
}

// ---------------------------------------------------------------------------
extern "C" void kernel_launch(void* const* d_in, const int* in_sizes, int n_in,
                              void* d_out, int out_size, void* d_ws,
                              size_t ws_size, hipStream_t stream) {
  const float* x_pos = (const float*)d_in[0];
  const float* fc0_w = (const float*)d_in[1];
  const float* fc0_b = (const float*)d_in[2];
  const float* w_ih  = (const float*)d_in[3];
  const float* w_hh  = (const float*)d_in[4];
  const float* b_ih  = (const float*)d_in[5];
  const float* b_hh  = (const float*)d_in[6];
  const float* fc1_w = (const float*)d_in[7];
  const float* fc1_b = (const float*)d_in[8];
  float* out = (float*)d_out;

  // Workspace layout (~188 MB):
  //   bf16 weights: fc0 1.1MB | w_ih 45MB | w_hh 45MB | fc1 11.3MB
  //   xb0/xb1 (T*B,H) bf16 ping-pong | gx (T*B,4H) f32 | cbuf (B,H) f32
  char* ws = (char*)d_ws;
  size_t off = 0;
  __bf16* wbf_fc0 = (__bf16*)(ws + off); off += (size_t)HDIM * HDIM * 2;
  __bf16* wbf_ih  = (__bf16*)(ws + off); off += (size_t)LAYERS * G4H * HDIM * 2;
  __bf16* wbf_hh  = (__bf16*)(ws + off); off += (size_t)LAYERS * G4H * HDIM * 2;
  __bf16* wbf_fc1 = (__bf16*)(ws + off); off += (size_t)HDIM * TSTEPS * HDIM * 2;
  size_t xbytes = (size_t)TSTEPS * BATCH * HDIM * 2;
  __bf16* xb[2];
  xb[0] = (__bf16*)(ws + off); off += xbytes;
  xb[1] = (__bf16*)(ws + off); off += xbytes;
  float* gx   = (float*)(ws + off); off += (size_t)TSTEPS * BATCH * G4H * 4;
  float* cbuf = (float*)(ws + off);

  const dim3 blk(256);
  const int M = TSTEPS * BATCH;  // 5120

  // 0) weight conversion fp32 -> bf16 (8 elems/thread)
  convert_w_kernel<<<(HDIM * HDIM) / 2048, blk, 0, stream>>>(fc0_w, wbf_fc0);
  convert_w_kernel<<<(LAYERS * G4H * HDIM) / 2048, blk, 0, stream>>>(w_ih, wbf_ih);
  convert_w_kernel<<<(LAYERS * G4H * HDIM) / 2048, blk, 0, stream>>>(w_hh, wbf_hh);
  convert_w_kernel<<<(HDIM * TSTEPS * HDIM) / 2048, blk, 0, stream>>>(fc1_w, wbf_fc1);

  // 1) (b,t) -> (t,b) transpose + bf16 convert
  reorder_x_kernel<<<(M * HDIM) / 256, blk, 0, stream>>>(x_pos, xb[0]);

  // 2) fc0: (5120,768) @ (768,768)^T + b -> bf16
  gemm_bias_kernel<<<dim3(M / 128, HDIM / 64), blk, 0, stream>>>(
      xb[0], wbf_fc0, fc0_b, nullptr, xb[1], M, HDIM, HDIM,
      /*rowStride=*/HDIM, /*chunkK=*/HDIM, /*chunkOuter=*/0, /*outBf16=*/1);

  int cur = 1;
  for (int l = 0; l < LAYERS; ++l) {
    const __bf16* wih_l = wbf_ih + (size_t)l * G4H * HDIM;
    const __bf16* whh_l = wbf_hh + (size_t)l * G4H * HDIM;

    // 3a) gx = X @ w_ih^T + b_ih + b_hh   (5120 x 3072, K=768) -> fp32
    gemm_bias_kernel<<<dim3(M / 128, G4H / 64), blk, 0, stream>>>(
        xb[cur], wih_l, b_ih + (size_t)l * G4H, b_hh + (size_t)l * G4H, gx,
        M, G4H, HDIM, HDIM, HDIM, 0, /*outBf16=*/0);

    // 3b) sequential scan: 10 fused recurrent-GEMM + gate kernels
    __bf16* ybuf = xb[1 - cur];
    for (int t = 0; t < TSTEPS; ++t) {
      lstm_step_kernel<<<dim3(BATCH / 128, HDIM / 16), blk, 0, stream>>>(
          gx, whh_l, ybuf, cbuf, t);
    }
    cur ^= 1;
  }

  // 4) fc1: (512, 7680) @ (7680, 768)^T + b -> fp32 out.
  //    A(m, k=w*H+h) gathered from ys[w][m][h] via chunked addressing.
  gemm_bias_kernel<<<dim3(BATCH / 128, HDIM / 64), blk, 0, stream>>>(
      xb[cur], wbf_fc1, fc1_b, nullptr, out, BATCH, HDIM, TSTEPS * HDIM,
      /*rowStride=*/HDIM, /*chunkK=*/HDIM,
      /*chunkOuter=*/(long)BATCH * HDIM, /*outBf16=*/0);
}
